// SwitchMoE_5832565588218
// MI455X (gfx1250) — compile-verified, matrix operand order checked by reading, not run on hardware
//
#include <hip/hip_runtime.h>
#include <hip/hip_bf16.h>
#include <math.h>

#define Ntok 8192
#define Dd   1024
#define Hh   4096
#define Ee   8
#define EPSv 1e-6f

#define TM 128
#define TN 256
#define TK 32
#define LDK 40   // LDS row stride in elements (80B) -> conflict-free, 16B aligned

typedef __attribute__((ext_vector_type(16))) __bf16 v16bf;
typedef __attribute__((ext_vector_type(8)))  float  v8f;
typedef __attribute__((ext_vector_type(8)))  unsigned short u16x8;

__device__ __forceinline__ unsigned short bfbits(float f) {
  return __builtin_bit_cast(unsigned short, (__bf16)f);
}

// Load a 16x32 bf16 fragment (A-layout per CDNA5 ISA 7.12.2) from an LDS
// sub-tile whose rows are LDK elements apart. B fragments use the identical
// layout because B-tiles are staged transposed (Bs[n][k]).
__device__ __forceinline__ v16bf load_frag(const __bf16* base, int lane) {
  const int m  = lane & 15;
  const int k0 = (lane >> 4) << 3;   // lanes 0-15: K 0..7/16..23; 16-31: 8..15/24..31
  const __bf16* p = base + m * LDK + k0;
  u16x8 lo = *(const u16x8*)(p);
  u16x8 hi = *(const u16x8*)(p + 16);
  v16bf a;
#pragma unroll
  for (int i = 0; i < 8; ++i) {
    a[i]     = __builtin_bit_cast(__bf16, (unsigned short)lo[i]);
    a[i + 8] = __builtin_bit_cast(__bf16, (unsigned short)hi[i]);
  }
  return a;
}

// ---------------------------------------------------------------- routing ---

__global__ void moe_init_kernel(int* counts, float* denom) {
  int t = threadIdx.x;
  if (t < Ee) { counts[t] = 0; denom[t] = 0.f; }
}

// Batched transpose + fp32->bf16: src[b][R][C] fp32 -> dst[b][C][R] bf16.
// 32x32 tiles, coalesced on both phases.
__global__ __launch_bounds__(256) void moe_tcvt_kernel(
    const float* __restrict__ src, __bf16* __restrict__ dst, int R, int C) {
  __shared__ __bf16 tile[32][33];
  const int b  = blockIdx.z;
  const int rt = blockIdx.y * 32;
  const int ct = blockIdx.x * 32;
  const int tid = threadIdx.x;
  const float* sp = src + ((size_t)b * R + rt) * C + ct;
#pragma unroll
  for (int i = 0; i < 4; ++i) {
    const int idx = tid + i * 256;           // 1024 elements
    const int r = idx >> 5, c = idx & 31;    // coalesced along c
    tile[r][c] = (__bf16)sp[(size_t)r * C + c];
  }
  __syncthreads();
  __bf16* dp = dst + ((size_t)b * C + ct) * R + rt;
#pragma unroll
  for (int i = 0; i < 2; ++i) {
    const int idx = tid + i * 256;           // 512 u32 pairs
    const int c = idx >> 4, rp = (idx & 15) * 2;  // coalesced along rp
    unsigned u = (unsigned)__builtin_bit_cast(unsigned short, tile[rp][c]) |
                 ((unsigned)__builtin_bit_cast(unsigned short, tile[rp + 1][c]) << 16);
    *(unsigned*)(dp + (size_t)c * R + rp) = u;
  }
}

// One wave32 per token: logits = x@wg + bg, softmax, top-1.
__global__ __launch_bounds__(256) void moe_gate_kernel(
    const float* __restrict__ x, const float* __restrict__ wg,
    const float* __restrict__ bg, int* __restrict__ top1,
    float* __restrict__ val, int* __restrict__ counts) {
  const int wv   = (blockIdx.x * blockDim.x + threadIdx.x) >> 5;  // token id
  const int lane = threadIdx.x & 31;
  if (wv >= Ntok) return;
  float acc[Ee];
#pragma unroll
  for (int e = 0; e < Ee; ++e) acc[e] = 0.f;
  const float* xr = x + (size_t)wv * Dd;
  for (int d = lane; d < Dd; d += 32) {
    const float xv = xr[d];
    const float* wr = wg + (size_t)d * Ee;
#pragma unroll
    for (int e = 0; e < Ee; ++e) acc[e] += xv * wr[e];
  }
#pragma unroll
  for (int e = 0; e < Ee; ++e)
#pragma unroll
    for (int off = 16; off > 0; off >>= 1)
      acc[e] += __shfl_xor(acc[e], off, 32);
  if (lane == 0) {
    float l[Ee];
#pragma unroll
    for (int e = 0; e < Ee; ++e) l[e] = acc[e] + bg[e];
    float mx = l[0]; int am = 0;
#pragma unroll
    for (int e = 1; e < Ee; ++e) if (l[e] > mx) { mx = l[e]; am = e; }  // first-max tie-break
    float s = 0.f;
#pragma unroll
    for (int e = 0; e < Ee; ++e) s += expf(l[e] - mx);
    top1[wv] = am;
    val[wv]  = 1.0f / s;                  // softmax prob of the argmax
    atomicAdd(&counts[am], 1);
  }
}

// Deterministic per-expert sum of masked scores (fixed partition + tree order).
__global__ __launch_bounds__(256) void moe_denom_kernel(
    const int* __restrict__ top1, const float* __restrict__ val,
    float* __restrict__ denom) {
  __shared__ float red[256];
  const int e = blockIdx.x;
  float s = 0.f;
  for (int n = threadIdx.x; n < Ntok; n += 256) s += (top1[n] == e) ? val[n] : 0.f;
  red[threadIdx.x] = s;
  __syncthreads();
  for (int o = 128; o > 0; o >>= 1) {
    if (threadIdx.x < o) red[threadIdx.x] += red[threadIdx.x + o];
    __syncthreads();
  }
  if (threadIdx.x == 0) denom[e] = red[0];
}

__global__ void moe_scan_kernel(const int* counts, int* offs, int* cursor) {
  if (threadIdx.x == 0) {
    int a = 0;
    for (int e = 0; e < Ee; ++e) { offs[e] = a; cursor[e] = a; a += counts[e]; }
    offs[Ee] = a;
  }
}

__global__ __launch_bounds__(256) void moe_assign_kernel(
    const int* __restrict__ top1, const float* __restrict__ val,
    const float* __restrict__ denom, int* __restrict__ cursor,
    int* __restrict__ tok_sorted, float* __restrict__ gsc) {
  const int n = blockIdx.x * 256 + threadIdx.x;
  if (n >= Ntok) return;
  const int e = top1[n];
  const int slot = atomicAdd(&cursor[e], 1);
  tok_sorted[slot] = n;
  gsc[slot] = val[n] / (denom[e] + EPSv) * (float)Ntok;   // capacity = N
}

// Gather tokens into expert-grouped order, converting fp32 -> bf16.
__global__ __launch_bounds__(256) void moe_gather_kernel(
    const float* __restrict__ x, const int* __restrict__ tok_sorted,
    __bf16* __restrict__ Xg) {
  const int g = blockIdx.x * 256 + threadIdx.x;  // one per 8 elements
  const int slot = g >> 7;                        // D/8 = 128 chunks per row
  const int col  = (g & 127) * 8;
  if (slot >= Ntok) return;
  const int tok = tok_sorted[slot];
  const float4* p = (const float4*)(x + (size_t)tok * Dd + col);
  float4 v0 = p[0], v1 = p[1];
  u16x8 o;
  o[0] = bfbits(v0.x); o[1] = bfbits(v0.y); o[2] = bfbits(v0.z); o[3] = bfbits(v0.w);
  o[4] = bfbits(v1.x); o[5] = bfbits(v1.y); o[6] = bfbits(v1.z); o[7] = bfbits(v1.w);
  *(u16x8*)(Xg + (size_t)slot * Dd + col) = o;
}

// -------------------------------------------------- pass A: H = gelu(XW1+b) --
// A = Xg (grouped tokens, bf16), B = w1t[e][h][d] (pre-transposed bf16).
// Tile 128x256, 8 waves as 2(M)x4(N), each wave a 64x64 patch = 16 WMMAs.

__global__ __launch_bounds__(256) void moe_ffn1_kernel(
    const __bf16* __restrict__ Xg, const __bf16* __restrict__ w1t,
    const float* __restrict__ b1, const int* __restrict__ offs,
    __bf16* __restrict__ Hs) {
  __shared__ alignas(16) __bf16 As[2][TM * LDK];   // 2 x 10240 B
  __shared__ alignas(16) __bf16 Bs[2][TN * LDK];   // 2 x 20480 B

  const int e    = blockIdx.z;
  const int base = offs[e];
  const int cnt  = offs[e + 1] - base;
  const int mt   = blockIdx.y;
  if (mt * TM >= cnt) return;                 // over-launched tile, uniform exit
  const int ho = blockIdx.x * TN;

  const int tid  = threadIdx.x;
  const int lane = tid & 31;
  const int w    = tid >> 5;
  const int wm   = (w & 1) * 64;              // 2 waves along M (64 rows each)
  const int wn   = (w >> 1) * 64;             // 4 waves along N (64 cols each)

  // A loader: 2 threads/row, 16 bf16 each
  const int ar = tid >> 1;
  const int ac = (tid & 1) * 16;
  int aslot = base + mt * TM + ar;
  if (aslot > Ntok - 1) aslot = Ntok - 1;     // clamp keeps OOB rows in-bounds
  const __bf16* Arow = Xg + (size_t)aslot * Dd;
  // B loader: 1 thread/row, 32 bf16 each
  const __bf16* Brow = w1t + ((size_t)e * Hh + ho + tid) * Dd;

  v8f acc[4][4] = {};
  u16x8 ra0, ra1, rb0, rb1, rb2, rb3;

  // preload tile 0
  ra0 = *(const u16x8*)(Arow + ac);
  ra1 = *(const u16x8*)(Arow + ac + 8);
  rb0 = *(const u16x8*)(Brow);
  rb1 = *(const u16x8*)(Brow + 8);
  rb2 = *(const u16x8*)(Brow + 16);
  rb3 = *(const u16x8*)(Brow + 24);
  *(u16x8*)&As[0][ar * LDK + ac]      = ra0;
  *(u16x8*)&As[0][ar * LDK + ac + 8]  = ra1;
  *(u16x8*)&Bs[0][tid * LDK]          = rb0;
  *(u16x8*)&Bs[0][tid * LDK + 8]      = rb1;
  *(u16x8*)&Bs[0][tid * LDK + 16]     = rb2;
  *(u16x8*)&Bs[0][tid * LDK + 24]     = rb3;
  __syncthreads();

  int cur = 0;
  for (int kb = 0; kb < Dd; kb += TK) {
    const int nxt = kb + TK;
    const bool more = nxt < Dd;
    if (more) {
      ra0 = *(const u16x8*)(Arow + nxt + ac);
      ra1 = *(const u16x8*)(Arow + nxt + ac + 8);
      rb0 = *(const u16x8*)(Brow + nxt);
      rb1 = *(const u16x8*)(Brow + nxt + 8);
      rb2 = *(const u16x8*)(Brow + nxt + 16);
      rb3 = *(const u16x8*)(Brow + nxt + 24);
    }
    v16bf bfr[4];
#pragma unroll
    for (int j = 0; j < 4; ++j) bfr[j] = load_frag(&Bs[cur][(wn + j * 16) * LDK], lane);
#pragma unroll
    for (int i = 0; i < 4; ++i) {
      v16bf af = load_frag(&As[cur][(wm + i * 16) * LDK], lane);
#pragma unroll
      for (int j = 0; j < 4; ++j)
        acc[i][j] = __builtin_amdgcn_wmma_f32_16x16x32_bf16(
            false, af, false, bfr[j], (short)0, acc[i][j], false, false);
    }
    if (more) {
      const int nbuf = cur ^ 1;
      *(u16x8*)&As[nbuf][ar * LDK + ac]      = ra0;
      *(u16x8*)&As[nbuf][ar * LDK + ac + 8]  = ra1;
      *(u16x8*)&Bs[nbuf][tid * LDK]          = rb0;
      *(u16x8*)&Bs[nbuf][tid * LDK + 8]      = rb1;
      *(u16x8*)&Bs[nbuf][tid * LDK + 16]     = rb2;
      *(u16x8*)&Bs[nbuf][tid * LDK + 24]     = rb3;
    }
    __syncthreads();
    cur ^= 1;
  }

  const int nl = lane & 15;
  const int mo = (lane >> 4) * 8;             // C layout: M = vgpr + 8*(lane>=16)
#pragma unroll
  for (int i = 0; i < 4; ++i) {
#pragma unroll
    for (int j = 0; j < 4; ++j) {
      const int col  = ho + wn + j * 16 + nl;
      const float bb = b1[(size_t)e * Hh + col];
#pragma unroll
      for (int v = 0; v < 8; ++v) {
        const int mrow = mt * TM + wm + i * 16 + mo + v;
        if (mrow < cnt) {
          float t = acc[i][j][v] + bb;
          t = 0.5f * t * (1.0f + erff(t * 0.70710678118654752f));   // exact-erf GELU
          Hs[(size_t)(base + mrow) * Hh + col] = (__bf16)t;
        }
      }
    }
  }
}

// ------------------------------------------ pass B: out = gate*(H W2 + b2) ---
// A = Hs (grouped, bf16), B = w2t[e][d][h] (pre-transposed bf16).

__global__ __launch_bounds__(256) void moe_ffn2_kernel(
    const __bf16* __restrict__ Hs, const __bf16* __restrict__ w2t,
    const float* __restrict__ b2, const int* __restrict__ offs,
    const int* __restrict__ tok_sorted, const float* __restrict__ gsc,
    float* __restrict__ out) {
  __shared__ alignas(16) __bf16 As[2][TM * LDK];
  __shared__ alignas(16) __bf16 Bs[2][TN * LDK];

  const int e    = blockIdx.z;
  const int base = offs[e];
  const int cnt  = offs[e + 1] - base;
  const int mt   = blockIdx.y;
  if (mt * TM >= cnt) return;
  const int dco = blockIdx.x * TN;

  const int tid  = threadIdx.x;
  const int lane = tid & 31;
  const int w    = tid >> 5;
  const int wm   = (w & 1) * 64;
  const int wn   = (w >> 1) * 64;

  const int ar = tid >> 1;
  const int ac = (tid & 1) * 16;
  int aslot = base + mt * TM + ar;
  if (aslot > Ntok - 1) aslot = Ntok - 1;
  const __bf16* Arow = Hs + (size_t)aslot * Hh;
  const __bf16* Brow = w2t + ((size_t)e * Dd + dco + tid) * Hh;

  v8f acc[4][4] = {};
  u16x8 ra0, ra1, rb0, rb1, rb2, rb3;

  ra0 = *(const u16x8*)(Arow + ac);
  ra1 = *(const u16x8*)(Arow + ac + 8);
  rb0 = *(const u16x8*)(Brow);
  rb1 = *(const u16x8*)(Brow + 8);
  rb2 = *(const u16x8*)(Brow + 16);
  rb3 = *(const u16x8*)(Brow + 24);
  *(u16x8*)&As[0][ar * LDK + ac]      = ra0;
  *(u16x8*)&As[0][ar * LDK + ac + 8]  = ra1;
  *(u16x8*)&Bs[0][tid * LDK]          = rb0;
  *(u16x8*)&Bs[0][tid * LDK + 8]      = rb1;
  *(u16x8*)&Bs[0][tid * LDK + 16]     = rb2;
  *(u16x8*)&Bs[0][tid * LDK + 24]     = rb3;
  __syncthreads();

  int cur = 0;
  for (int kb = 0; kb < Hh; kb += TK) {
    const int nxt = kb + TK;
    const bool more = nxt < Hh;
    if (more) {
      ra0 = *(const u16x8*)(Arow + nxt + ac);
      ra1 = *(const u16x8*)(Arow + nxt + ac + 8);
      rb0 = *(const u16x8*)(Brow + nxt);
      rb1 = *(const u16x8*)(Brow + nxt + 8);
      rb2 = *(const u16x8*)(Brow + nxt + 16);
      rb3 = *(const u16x8*)(Brow + nxt + 24);
    }
    v16bf bfr[4];
#pragma unroll
    for (int j = 0; j < 4; ++j) bfr[j] = load_frag(&Bs[cur][(wn + j * 16) * LDK], lane);
#pragma unroll
    for (int i = 0; i < 4; ++i) {
      v16bf af = load_frag(&As[cur][(wm + i * 16) * LDK], lane);
#pragma unroll
      for (int j = 0; j < 4; ++j)
        acc[i][j] = __builtin_amdgcn_wmma_f32_16x16x32_bf16(
            false, af, false, bfr[j], (short)0, acc[i][j], false, false);
    }
    if (more) {
      const int nbuf = cur ^ 1;
      *(u16x8*)&As[nbuf][ar * LDK + ac]      = ra0;
      *(u16x8*)&As[nbuf][ar * LDK + ac + 8]  = ra1;
      *(u16x8*)&Bs[nbuf][tid * LDK]          = rb0;
      *(u16x8*)&Bs[nbuf][tid * LDK + 8]      = rb1;
      *(u16x8*)&Bs[nbuf][tid * LDK + 16]     = rb2;
      *(u16x8*)&Bs[nbuf][tid * LDK + 24]     = rb3;
    }
    __syncthreads();
    cur ^= 1;
  }

  const int nl = lane & 15;
  const int mo = (lane >> 4) * 8;
#pragma unroll
  for (int i = 0; i < 4; ++i) {
#pragma unroll
    for (int j = 0; j < 4; ++j) {
      const int col  = dco + wn + j * 16 + nl;
      const float bb = b2[(size_t)e * Dd + col];
#pragma unroll
      for (int v = 0; v < 8; ++v) {
        const int mrow = mt * TM + wm + i * 16 + mo + v;
        if (mrow < cnt) {
          const int tok = tok_sorted[base + mrow];
          const float g = gsc[base + mrow];
          out[(size_t)tok * Dd + col] = g * (acc[i][j][v] + bb);
        }
      }
    }
  }
}

// ------------------------------------------------------------------ launch ---

extern "C" void kernel_launch(void* const* d_in, const int* in_sizes, int n_in,
                              void* d_out, int out_size, void* d_ws, size_t ws_size,
                              hipStream_t stream) {
  (void)in_sizes; (void)n_in; (void)out_size; (void)ws_size;
  const float* x  = (const float*)d_in[0];
  const float* wg = (const float*)d_in[1];
  const float* bg = (const float*)d_in[2];
  const float* w1 = (const float*)d_in[3];
  const float* b1 = (const float*)d_in[4];
  const float* w2 = (const float*)d_in[5];
  const float* b2 = (const float*)d_in[6];
  float* out = (float*)d_out;

  char* ws = (char*)d_ws;
  int*    top1       = (int*)(ws);                       // N ints
  float*  val        = (float*)(ws + 32768);             // N floats
  int*    counts     = (int*)(ws + 65536);               // E
  int*    offs       = (int*)(ws + 65536 + 64);          // E+1
  int*    cursor     = (int*)(ws + 65536 + 128);         // E
  float*  denom      = (float*)(ws + 65536 + 192);       // E
  int*    tok_sorted = (int*)(ws + 65792);               // N ints
  float*  gsc        = (float*)(ws + 98560);             // N floats
  __bf16* Xg         = (__bf16*)(ws + 131328);           // N*D bf16  (16 MB)
  __bf16* Hs         = (__bf16*)(ws + 131328 + (size_t)Ntok * Dd * 2);          // N*H bf16 (64 MB)
  __bf16* w1t        = (__bf16*)((char*)Hs + (size_t)Ntok * Hh * 2);            // E*[H][D] bf16 (64 MB)
  __bf16* w2t        = (__bf16*)((char*)w1t + (size_t)Ee * Dd * Hh * 2);        // E*[D][H] bf16 (64 MB)

  moe_init_kernel<<<1, 32, 0, stream>>>(counts, denom);
  // w1[e][D][H] -> w1t[e][H][D]; w2[e][H][D] -> w2t[e][D][H]
  moe_tcvt_kernel<<<dim3(Hh / 32, Dd / 32, Ee), 256, 0, stream>>>(w1, w1t, Dd, Hh);
  moe_tcvt_kernel<<<dim3(Dd / 32, Hh / 32, Ee), 256, 0, stream>>>(w2, w2t, Hh, Dd);
  moe_gate_kernel<<<Ntok / 8, 256, 0, stream>>>(x, wg, bg, top1, val, counts);
  moe_denom_kernel<<<Ee, 256, 0, stream>>>(top1, val, denom);
  moe_scan_kernel<<<1, 32, 0, stream>>>(counts, offs, cursor);
  moe_assign_kernel<<<Ntok / 256, 256, 0, stream>>>(top1, val, denom, cursor, tok_sorted, gsc);
  moe_gather_kernel<<<(Ntok * (Dd / 8)) / 256, 256, 0, stream>>>(x, tok_sorted, Xg);
  moe_ffn1_kernel<<<dim3(Hh / TN, Ntok / TM, Ee), 256, 0, stream>>>(Xg, w1t, b1, offs, Hs);
  moe_ffn2_kernel<<<dim3(Dd / TN, Ntok / TM, Ee), 256, 0, stream>>>(Hs, w2t, b2, offs, tok_sorted, gsc, out);
}
